// Step3TextDecoderLayer_67808943669367
// MI455X (gfx1250) — compile-verified, matrix operand order checked by reading, not run on hardware
//
#include <hip/hip_runtime.h>
#include <hip/hip_bf16.h>
#include <cstdint>

// ---------------- constants (match reference) ----------------
#define HH   1024
#define NH   16
#define DD   128
#define QQ   512
#define EE   32
#define II   1024
#define TOPK 3
#define SS   2048
#define QKVW (QQ + 2*DD)   // 768
#define NEGV (-1e30f)

typedef __attribute__((ext_vector_type(16))) __bf16 v16bf;
typedef __attribute__((ext_vector_type(8)))  __bf16 v8bf;
typedef __attribute__((ext_vector_type(8)))  float  v8f;

// native RNE convert -> lets backend use hardware cvt ops
__device__ __forceinline__ __bf16 f2bf(float f) { return (__bf16)f; }

__device__ __forceinline__ v8f wmma_bf16(v16bf a, v16bf b, v8f c) {
    return __builtin_amdgcn_wmma_f32_16x16x32_bf16(false, a, false, b, (short)0, c, false, false);
}

// Fragment loader: rowbase -> this lane's row, 32-wide K window.
// elements e<8 -> k = 8*half+e ; e>=8 -> k = 16+8*half+(e-8)
__device__ __forceinline__ v16bf load_frag(const __bf16* rowbase, int half) {
    const v8bf* p = (const v8bf*)rowbase;
    v8bf lo = p[half];
    v8bf hi = p[2 + half];
    return __builtin_shufflevector(lo, hi, 0,1,2,3,4,5,6,7,8,9,10,11,12,13,14,15);
}

// ---- CDNA5 async global->LDS path (guarded: compiles either way) ----
#if defined(__has_builtin)
#  if __has_builtin(__builtin_amdgcn_global_load_async_to_lds_b128)
#    define HAVE_ASYNC 1
#  endif
#endif

#if defined(HAVE_ASYNC)
typedef int v4i_vs __attribute__((vector_size(16)));
__device__ __forceinline__ void async_ld16(void* lds, const void* g) {
    // param1: AS(1) ("__device__") v4i*, non-const (per compiler diagnostic)
    auto gp = (__attribute__((address_space(1))) v4i_vs*)(g);
    // param2: AS(3) ("__shared__") v4i* built from the 32-bit LDS offset
    auto lp = (__attribute__((address_space(3))) v4i_vs*)(unsigned int)(unsigned long long)(lds);
    __builtin_amdgcn_global_load_async_to_lds_b128(gp, lp, 0, 0);
}
#  if __has_builtin(__builtin_amdgcn_s_wait_asynccnt)
#    define WAIT_ASYNC() __builtin_amdgcn_s_wait_asynccnt(0)
#  else
#    define WAIT_ASYNC()
#  endif
#else
#  define WAIT_ASYNC()
#endif

#define LDA_P 40   // 32 + 8 pad, keeps 16B-aligned per-lane v8bf loads

// ============================================================
// Dense bf16-WMMA GEMM, register-staged software pipeline.
// C[M,N] = A[M,K] @ B[K,N] (+resid). Tile 128x128x32, 8 waves.
// ============================================================
template<bool GUARD>
__global__ __launch_bounds__(256)
void gemm_bf16_kernel(const float* __restrict__ A, int lda,
                      const float* __restrict__ B, int ldb,
                      float* __restrict__ C, int ldc,
                      int M, int N, int K,
                      const float* __restrict__ resid)
{
    __shared__ __bf16 Alds[128 * LDA_P];
    __shared__ __bf16 Blds[128 * LDA_P];   // transposed: [n][k]

    const int tid = threadIdx.x;
    const int wid = tid >> 5, lane = tid & 31;
    const int half = lane >> 4, l16 = lane & 15;
    const int wm = wid & 1, wn = wid >> 1;
    const int m0 = blockIdx.y * 128;
    const int n0 = blockIdx.x * 128;

    v8f acc[4][2];
#pragma unroll
    for (int i = 0; i < 4; i++)
#pragma unroll
        for (int j = 0; j < 2; j++) acc[i][j] = {};

    float4 ra[4], rb[4];

    auto loadA = [&](int k0) {
#pragma unroll
        for (int i = 0; i < 4; i++) {
            int c = tid + i * 256;                // 1024 float4 chunks of A tile
            int r = c >> 3, kc = (c & 7) * 4;
            int gr = m0 + r;
            if (GUARD && gr >= M) ra[i] = make_float4(0.f, 0.f, 0.f, 0.f);
            else ra[i] = *(const float4*)&A[(size_t)gr * lda + k0 + kc];
        }
    };
    auto loadB = [&](int k0) {
#pragma unroll
        for (int i = 0; i < 4; i++) {
            int c = tid + i * 256;
            int kk = c >> 5, n4 = (c & 31) * 4;
            int gn = n0 + n4;
            if (GUARD && gn >= N) rb[i] = make_float4(0.f, 0.f, 0.f, 0.f);
            else rb[i] = *(const float4*)&B[(size_t)(k0 + kk) * ldb + gn];
        }
    };
    auto commit = [&]() {
#pragma unroll
        for (int i = 0; i < 4; i++) {
            int c = tid + i * 256;
            int r = c >> 3, kc = (c & 7) * 4;
            __bf16* d = &Alds[r * LDA_P + kc];
            d[0] = f2bf(ra[i].x); d[1] = f2bf(ra[i].y);
            d[2] = f2bf(ra[i].z); d[3] = f2bf(ra[i].w);
        }
#pragma unroll
        for (int i = 0; i < 4; i++) {
            int c = tid + i * 256;
            int kk = c >> 5, n4 = (c & 31) * 4;
            Blds[(n4 + 0) * LDA_P + kk] = f2bf(rb[i].x);
            Blds[(n4 + 1) * LDA_P + kk] = f2bf(rb[i].y);
            Blds[(n4 + 2) * LDA_P + kk] = f2bf(rb[i].z);
            Blds[(n4 + 3) * LDA_P + kk] = f2bf(rb[i].w);
        }
    };

    loadA(0); loadB(0);
    for (int k0 = 0; k0 < K; k0 += 32) {
        commit();
        __syncthreads();
        if (k0 + 32 < K) { loadA(k0 + 32); loadB(k0 + 32); }   // fly under the WMMAs

        v16bf bf[2];
#pragma unroll
        for (int cn = 0; cn < 2; cn++)
            bf[cn] = load_frag(&Blds[(wn * 32 + cn * 16 + l16) * LDA_P], half);
#pragma unroll
        for (int cm = 0; cm < 4; cm++) {
            v16bf af = load_frag(&Alds[(wm * 64 + cm * 16 + l16) * LDA_P], half);
#pragma unroll
            for (int cn = 0; cn < 2; cn++)
                acc[cm][cn] = wmma_bf16(af, bf[cn], acc[cm][cn]);
        }
        __syncthreads();
    }

#pragma unroll
    for (int cm = 0; cm < 4; cm++)
#pragma unroll
        for (int cn = 0; cn < 2; cn++)
#pragma unroll
            for (int v = 0; v < 8; v++) {
                int m = m0 + wm * 64 + cm * 16 + v + 8 * half;
                int n = n0 + wn * 32 + cn * 16 + l16;
                if (!GUARD || (m < M && n < N)) {
                    float r = resid ? resid[(size_t)m * ldc + n] : 0.f;
                    C[(size_t)m * ldc + n] = acc[cm][cn][v] + r;
                }
            }
}
template __global__ void gemm_bf16_kernel<false>(const float*, int, const float*, int,
                                                 float*, int, int, int, int, const float*);

// ============================================================
// GLU GEMM: act = silu(X@Bg)*(X@Bu).  Optional MoE gather.
// Output: f32 (Actf) or bf16 (Actb) — exactly one non-null.
// ============================================================
__global__ __launch_bounds__(256)
void glu_gemm_kernel(const float* __restrict__ X, int ldx,
                     const float* __restrict__ Bgu, size_t estride, int ldb,
                     float* __restrict__ Actf, __bf16* __restrict__ Actb, int ldact,
                     int K, int Ihalf, int Mplain,
                     const int* __restrict__ seg_off, const int* __restrict__ seg_cnt,
                     const int* __restrict__ assign_tok)
{
    int eM = Mplain, base = 0;
    const float* Bp = Bgu;
    if (seg_off) {
        int e = blockIdx.z;
        eM = seg_cnt[e]; base = seg_off[e];
        Bp = Bgu + estride * (size_t)e;
        if ((int)(blockIdx.y * 128) >= eM) return;
    }

    __shared__ __bf16 Alds[128 * LDA_P];
    __shared__ __bf16 Bg[128 * LDA_P];
    __shared__ __bf16 Bu[128 * LDA_P];

    const int tid = threadIdx.x;
    const int wid = tid >> 5, lane = tid & 31;
    const int half = lane >> 4, l16 = lane & 15;
    const int wm = wid & 1, wn = wid >> 1;
    const int m0 = blockIdx.y * 128;
    const int n0 = blockIdx.x * 128;

    v8f ag[4][2], au[4][2];
#pragma unroll
    for (int i = 0; i < 4; i++)
#pragma unroll
        for (int j = 0; j < 2; j++) { ag[i][j] = {}; au[i][j] = {}; }

    for (int k0 = 0; k0 < K; k0 += 32) {
#pragma unroll
        for (int i = 0; i < 4; i++) {
            int c = tid + i * 256;
            int r = c >> 3, kc = (c & 7) * 4;
            int gr = m0 + r;
            float4 v = make_float4(0.f, 0.f, 0.f, 0.f);
            if (gr < eM) {
                int arow = seg_off ? assign_tok[base + gr] : gr;
                v = *(const float4*)&X[(size_t)arow * ldx + k0 + kc];
            }
            __bf16* d = &Alds[r * LDA_P + kc];
            d[0] = f2bf(v.x); d[1] = f2bf(v.y); d[2] = f2bf(v.z); d[3] = f2bf(v.w);
        }
#pragma unroll
        for (int i = 0; i < 4; i++) {
            int c = tid + i * 256;
            int kk = c >> 5, n4 = (c & 31) * 4;
            const float* src = &Bp[(size_t)(k0 + kk) * ldb + n0 + n4];
            float4 vg = *(const float4*)src;
            float4 vu = *(const float4*)(src + Ihalf);
            Bg[(n4 + 0) * LDA_P + kk] = f2bf(vg.x); Bg[(n4 + 1) * LDA_P + kk] = f2bf(vg.y);
            Bg[(n4 + 2) * LDA_P + kk] = f2bf(vg.z); Bg[(n4 + 3) * LDA_P + kk] = f2bf(vg.w);
            Bu[(n4 + 0) * LDA_P + kk] = f2bf(vu.x); Bu[(n4 + 1) * LDA_P + kk] = f2bf(vu.y);
            Bu[(n4 + 2) * LDA_P + kk] = f2bf(vu.z); Bu[(n4 + 3) * LDA_P + kk] = f2bf(vu.w);
        }
        __syncthreads();

        if (k0 + 32 < K)
            __builtin_prefetch(&Bp[(size_t)(k0 + 32 + (tid & 31)) * ldb + n0], 0, 1);

        v16bf bgf[2], buf_[2];
#pragma unroll
        for (int cn = 0; cn < 2; cn++) {
            bgf[cn]  = load_frag(&Bg[(wn * 32 + cn * 16 + l16) * LDA_P], half);
            buf_[cn] = load_frag(&Bu[(wn * 32 + cn * 16 + l16) * LDA_P], half);
        }
#pragma unroll
        for (int cm = 0; cm < 4; cm++) {
            v16bf af = load_frag(&Alds[(wm * 64 + cm * 16 + l16) * LDA_P], half);
#pragma unroll
            for (int cn = 0; cn < 2; cn++) {
                ag[cm][cn] = wmma_bf16(af, bgf[cn],  ag[cm][cn]);
                au[cm][cn] = wmma_bf16(af, buf_[cn], au[cm][cn]);
            }
        }
        __syncthreads();
    }

#pragma unroll
    for (int cm = 0; cm < 4; cm++)
#pragma unroll
        for (int cn = 0; cn < 2; cn++)
#pragma unroll
            for (int v = 0; v < 8; v++) {
                int m = m0 + wm * 64 + cm * 16 + v + 8 * half;
                int n = n0 + wn * 32 + cn * 16 + l16;
                if (m < eM) {
                    float g = ag[cm][cn][v], u = au[cm][cn][v];
                    float r = (g / (1.f + __expf(-g))) * u;
                    if (Actf) Actf[(size_t)(base + m) * ldact + n] = r;
                    else      Actb[(size_t)(base + m) * ldact + n] = f2bf(r);
                }
            }
}

// ============================================================
// MoE down-projection: moe[tok] += w * (act_row @ w_down_e[e])
// A is bf16 in global (written by GLU) -> raw async copy to LDS.
// ============================================================
__global__ __launch_bounds__(256)
void down_moe_kernel(const __bf16* __restrict__ Act, int ldact,
                     const float* __restrict__ Bd, size_t estride, int ldb,
                     float* __restrict__ Out, int ldout, int K,
                     const int* __restrict__ seg_off, const int* __restrict__ seg_cnt,
                     const int* __restrict__ assign_tok, const float* __restrict__ assign_w)
{
    int e = blockIdx.z;
    int eM = seg_cnt[e], base = seg_off[e];
    if ((int)(blockIdx.y * 128) >= eM) return;
    const float* Bp = Bd + estride * (size_t)e;

    __shared__ __bf16 Alds[128 * LDA_P];
    __shared__ __bf16 Blds[128 * LDA_P];

    const int tid = threadIdx.x;
    const int wid = tid >> 5, lane = tid & 31;
    const int half = lane >> 4, l16 = lane & 15;
    const int wm = wid & 1, wn = wid >> 1;
    const int m0 = blockIdx.y * 128;
    const int n0 = blockIdx.x * 128;

    v8f acc[4][2];
#pragma unroll
    for (int i = 0; i < 4; i++)
#pragma unroll
        for (int j = 0; j < 2; j++) acc[i][j] = {};

    for (int k0 = 0; k0 < K; k0 += 32) {
        // ---- A: 128x32 bf16 = 512 x 16B chunks (raw copy, no convert) ----
#pragma unroll
        for (int i = 0; i < 2; i++) {
            int c = tid + i * 256;
            int r = c >> 2, kc8 = (c & 3) * 8;
            int gr = m0 + r;
            void* ldst = &Alds[r * LDA_P + kc8];
            if (gr < eM) {
                const void* g = &Act[(size_t)(base + gr) * ldact + k0 + kc8];
#if defined(HAVE_ASYNC)
                async_ld16(ldst, g);                  // global_load_async_to_lds_b128
#else
                *(uint4*)ldst = *(const uint4*)g;
#endif
            } else {
                *(uint4*)ldst = make_uint4(0u, 0u, 0u, 0u);
            }
        }
        // ---- B: f32 -> bf16, transposed [n][k] ----
#pragma unroll
        for (int i = 0; i < 4; i++) {
            int c = tid + i * 256;
            int kk = c >> 5, n4 = (c & 31) * 4;
            float4 v = *(const float4*)&Bp[(size_t)(k0 + kk) * ldb + n0 + n4];
            Blds[(n4 + 0) * LDA_P + kk] = f2bf(v.x); Blds[(n4 + 1) * LDA_P + kk] = f2bf(v.y);
            Blds[(n4 + 2) * LDA_P + kk] = f2bf(v.z); Blds[(n4 + 3) * LDA_P + kk] = f2bf(v.w);
        }
        WAIT_ASYNC();          // s_wait_asynccnt 0 (this wave's copies)
        __syncthreads();       // all waves' tiles visible

        v16bf bf[2];
#pragma unroll
        for (int cn = 0; cn < 2; cn++)
            bf[cn] = load_frag(&Blds[(wn * 32 + cn * 16 + l16) * LDA_P], half);
#pragma unroll
        for (int cm = 0; cm < 4; cm++) {
            v16bf af = load_frag(&Alds[(wm * 64 + cm * 16 + l16) * LDA_P], half);
#pragma unroll
            for (int cn = 0; cn < 2; cn++)
                acc[cm][cn] = wmma_bf16(af, bf[cn], acc[cm][cn]);
        }
        __syncthreads();
    }

#pragma unroll
    for (int cm = 0; cm < 4; cm++)
#pragma unroll
        for (int cn = 0; cn < 2; cn++)
#pragma unroll
            for (int v = 0; v < 8; v++) {
                int m = m0 + wm * 64 + cm * 16 + v + 8 * half;
                int n = n0 + wn * 32 + cn * 16 + l16;
                if (m < eM) {
                    int tok = assign_tok[base + m];
                    float w = assign_w[base + m];
                    atomicAdd(&Out[(size_t)tok * ldout + n], w * acc[cm][cn][v]);
                }
            }
}

// ============================================================
// Flash attention
// ============================================================
#define KP 72
#define DP 136

__global__ __launch_bounds__(256)
void attn_kernel(const float* __restrict__ qp,
                 const float* __restrict__ qkv,
                 float* __restrict__ outp)
{
    __shared__ __bf16 Klds[64 * DP];
    __shared__ __bf16 Vlds[128 * KP];
    __shared__ __bf16 Plds[8 * 16 * KP];

    const int tid = threadIdx.x;
    const int wid = tid >> 5, lane = tid & 31;
    const int half = lane >> 4, l16 = lane & 15;
    const int h = blockIdx.y;
    const int qbase = blockIdx.x * 128;
    const int qrow_a = qbase + wid * 16 + l16;
    const float scale = 0.088388347648318447f;

    v16bf qf[4];
#pragma unroll
    for (int kd = 0; kd < 4; kd++) {
        v16bf x;
#pragma unroll
        for (int e = 0; e < 16; e++) {
            int d = kd * 32 + ((e < 8) ? 0 : 16) + 8 * half + (e & 7);
            x[e] = f2bf(qp[(size_t)qrow_a * (NH * DD) + h * DD + d] * scale);
        }
        qf[kd] = x;
    }

    v8f oacc[8];
#pragma unroll
    for (int i = 0; i < 8; i++) oacc[i] = {};
    float rmax[8], rsum[8];
#pragma unroll
    for (int v = 0; v < 8; v++) { rmax[v] = -3e38f; rsum[v] = 0.f; }

    const int ntiles = 2 * blockIdx.x + 2;
    for (int kt = 0; kt < ntiles; kt++) {
        const int kbase = kt * 64;
        for (int i = tid; i < 64 * 32; i += 256) {
            int r = i >> 5, c4 = (i & 31) * 4;
            float4 v = *(const float4*)&qkv[(size_t)(kbase + r) * QKVW + QQ + c4];
            __bf16* d = &Klds[r * DP + c4];
            d[0] = f2bf(v.x); d[1] = f2bf(v.y); d[2] = f2bf(v.z); d[3] = f2bf(v.w);
        }
        for (int i = tid; i < 64 * 32; i += 256) {
            int r = i >> 5, c4 = (i & 31) * 4;
            float4 v = *(const float4*)&qkv[(size_t)(kbase + r) * QKVW + QQ + DD + c4];
            Vlds[(c4 + 0) * KP + r] = f2bf(v.x);
            Vlds[(c4 + 1) * KP + r] = f2bf(v.y);
            Vlds[(c4 + 2) * KP + r] = f2bf(v.z);
            Vlds[(c4 + 3) * KP + r] = f2bf(v.w);
        }
        __syncthreads();

        v8f sc[4];
#pragma unroll
        for (int ns = 0; ns < 4; ns++) {
            sc[ns] = {};
#pragma unroll
            for (int kd = 0; kd < 4; kd++) {
                v16bf b = load_frag(&Klds[(ns * 16 + l16) * DP + kd * 32], half);
                sc[ns] = wmma_bf16(qf[kd], b, sc[ns]);
            }
        }

#pragma unroll
        for (int ns = 0; ns < 4; ns++)
#pragma unroll
            for (int v = 0; v < 8; v++) {
                int key = kbase + ns * 16 + l16;
                int qr  = qbase + wid * 16 + v + 8 * half;
                if (key > qr) sc[ns][v] = NEGV;
            }
#pragma unroll
        for (int v = 0; v < 8; v++) {
            float tm = sc[0][v];
#pragma unroll
            for (int ns = 1; ns < 4; ns++) tm = fmaxf(tm, sc[ns][v]);
#pragma unroll
            for (int m = 1; m < 16; m <<= 1) tm = fmaxf(tm, __shfl_xor(tm, m, 32));
            float nm = fmaxf(rmax[v], tm);
            float al = __expf(rmax[v] - nm);
            rmax[v] = nm;
            float ps = 0.f;
#pragma unroll
            for (int ns = 0; ns < 4; ns++) {
                float p = __expf(sc[ns][v] - nm);
                sc[ns][v] = p;
                ps += p;
            }
#pragma unroll
            for (int m = 1; m < 16; m <<= 1) ps += __shfl_xor(ps, m, 32);
            rsum[v] = rsum[v] * al + ps;
#pragma unroll
            for (int ds = 0; ds < 8; ds++) oacc[ds][v] *= al;
        }

#pragma unroll
        for (int ns = 0; ns < 4; ns++)
#pragma unroll
            for (int v = 0; v < 8; v++)
                Plds[wid * 16 * KP + (v + 8 * half) * KP + ns * 16 + l16] = f2bf(sc[ns][v]);

#pragma unroll
        for (int kc = 0; kc < 2; kc++) {
            v16bf a = load_frag(&Plds[wid * 16 * KP + l16 * KP + kc * 32], half);
#pragma unroll
            for (int ds = 0; ds < 8; ds++) {
                v16bf b = load_frag(&Vlds[(ds * 16 + l16) * KP + kc * 32], half);
                oacc[ds] = wmma_bf16(a, b, oacc[ds]);
            }
        }
        __syncthreads();
    }

#pragma unroll
    for (int v = 0; v < 8; v++) {
        float inv = 1.f / rsum[v];
        int qr = qbase + wid * 16 + v + 8 * half;
#pragma unroll
        for (int ds = 0; ds < 8; ds++)
            outp[(size_t)qr * (NH * DD) + h * DD + ds * 16 + l16] = oacc[ds][v] * inv;
    }
}

// ============================================================ small kernels
__global__ void rmsnorm_kernel(const float* __restrict__ in, int in_stride, int in_off,
                               const float* __restrict__ w,
                               float* __restrict__ out, int out_stride, int ncols)
{
    __shared__ float red[256];
    int row = blockIdx.x, tid = threadIdx.x;
    const float* x = in + (size_t)row * in_stride + in_off;
    float s = 0.f;
    for (int c = tid; c < ncols; c += 256) { float v = x[c]; s += v * v; }
    red[tid] = s; __syncthreads();
    for (int off = 128; off > 0; off >>= 1) { if (tid < off) red[tid] += red[tid + off]; __syncthreads(); }
    float rs = rsqrtf(red[0] / (float)ncols + 1e-5f);
    for (int c = tid; c < ncols; c += 256)
        out[(size_t)row * out_stride + c] = x[c] * rs * w[c];
}

__global__ void rope_kernel(float* __restrict__ buf, int row_stride, int col_off,
                            int rows_per_tok)
{
    int row = blockIdx.x, j = threadIdx.x;
    int t = row / rows_per_tok;
    float inv = __powf(500000.f, -((float)(2 * j)) / (float)DD);
    float f = (float)t * inv, s, c;
    __sincosf(f, &s, &c);
    float* p = buf + (size_t)row * row_stride + col_off;
    float x1 = p[j], x2 = p[j + DD / 2];
    p[j]          = x1 * c - x2 * s;
    p[j + DD / 2] = x2 * c + x1 * s;
}

__global__ void router_kernel(const float* __restrict__ x, const float* __restrict__ wr,
                              float* __restrict__ topw, int* __restrict__ topi)
{
    int wid = threadIdx.x >> 5, lane = threadIdx.x & 31;
    int t = blockIdx.x * 4 + wid;
    if (t >= SS) return;
    float l = 0.f;
    for (int hh = 0; hh < HH; hh++) l += x[(size_t)t * HH + hh] * wr[hh * EE + lane];
    float m = l;
#pragma unroll
    for (int xm = 1; xm < 32; xm <<= 1) m = fmaxf(m, __shfl_xor(m, xm, 32));
    float p = __expf(l - m), s = p;
#pragma unroll
    for (int xm = 1; xm < 32; xm <<= 1) s += __shfl_xor(s, xm, 32);
    p /= s;
    float pm = p, wsum = 0.f, myw[TOPK]; int myi[TOPK];
#pragma unroll
    for (int j = 0; j < TOPK; j++) {
        float mv = pm;
#pragma unroll
        for (int xm = 1; xm < 32; xm <<= 1) mv = fmaxf(mv, __shfl_xor(mv, xm, 32));
        unsigned long long b = __ballot(pm == mv);
        int idx = __ffsll(b) - 1;
        myi[j] = idx; myw[j] = mv; wsum += mv;
        if (lane == idx) pm = -1.f;
    }
    if (lane == 0)
        for (int j = 0; j < TOPK; j++) { topi[t * TOPK + j] = myi[j]; topw[t * TOPK + j] = myw[j] / wsum; }
}

__global__ void route_build_kernel(const int* __restrict__ topi, const float* __restrict__ topw,
                                   int* __restrict__ seg_off, int* __restrict__ seg_cnt,
                                   int* __restrict__ assign_tok, float* __restrict__ assign_w)
{
    __shared__ int cnt[EE], off[EE], fill[EE];
    int tid = threadIdx.x;
    if (tid < EE) cnt[tid] = 0;
    __syncthreads();
    for (int a = tid; a < SS * TOPK; a += 256) atomicAdd(&cnt[topi[a]], 1);
    __syncthreads();
    if (tid == 0) { int s = 0; for (int e = 0; e < EE; e++) { off[e] = s; s += cnt[e]; } }
    __syncthreads();
    if (tid < EE) { seg_off[tid] = off[tid]; seg_cnt[tid] = cnt[tid]; fill[tid] = 0; }
    __syncthreads();
    for (int a = tid; a < SS * TOPK; a += 256) {
        int e = topi[a];
        int pos = off[e] + atomicAdd(&fill[e], 1);
        assign_tok[pos] = a / TOPK;
        assign_w[pos]   = topw[a];
    }
}

__global__ void zero_kernel(float* __restrict__ p, int n) {
    int i = blockIdx.x * blockDim.x + threadIdx.x;
    if (i < n) p[i] = 0.f;
}

__global__ void final_add_kernel(const float* __restrict__ a, const float* __restrict__ b,
                                 const float* __restrict__ c, float* __restrict__ o, int n) {
    int i = blockIdx.x * blockDim.x + threadIdx.x;
    if (i < n) o[i] = a[i] + b[i] + c[i];
}

// ============================================================ launcher
extern "C" void kernel_launch(void* const* d_in, const int* in_sizes, int n_in,
                              void* d_out, int out_size, void* d_ws, size_t ws_size,
                              hipStream_t stream)
{
    const float* hidden   = (const float*)d_in[0];
    const float* w_in     = (const float*)d_in[1];
    const float* w_qkv    = (const float*)d_in[2];
    const float* w_inter  = (const float*)d_in[3];
    const float* w_q      = (const float*)d_in[4];
    const float* w_o      = (const float*)d_in[5];
    const float* w_post   = (const float*)d_in[6];
    const float* w_gu_se  = (const float*)d_in[7];
    const float* w_dn_se  = (const float*)d_in[8];
    const float* w_router = (const float*)d_in[9];
    const float* w_gu_e   = (const float*)d_in[10];
    const float* w_dn_e   = (const float*)d_in[11];
    float* out = (float*)d_out;

    float* f = (float*)d_ws;
    size_t o = 0;
    float* h_norm   = f + o; o += (size_t)SS * HH;
    float* qkv      = f + o; o += (size_t)SS * QKVW;
    float* q_rms    = f + o; o += (size_t)SS * QQ;
    float* q_proj   = f + o; o += (size_t)SS * NH * DD;
    float* attn_out = f + o; o += (size_t)SS * NH * DD;
    float* resid2   = f + o; o += (size_t)SS * HH;
    float* h_post   = f + o; o += (size_t)SS * HH;
    float* act_se   = f + o; o += (size_t)SS * II;
    float* share    = f + o; o += (size_t)SS * HH;
    float* moe      = f + o; o += (size_t)SS * HH;
    float* topw     = f + o; o += (size_t)SS * TOPK;
    float* assign_w = f + o; o += (size_t)SS * TOPK;
    __bf16* act_e   = (__bf16*)(f + o); o += (size_t)SS * TOPK * II / 2 + 64;
    int* ip = (int*)(f + o);
    int* topi       = ip;            ip += SS * TOPK;
    int* seg_off    = ip;            ip += EE;
    int* seg_cnt    = ip;            ip += EE;
    int* assign_tok = ip;            ip += SS * TOPK;

    rmsnorm_kernel<<<SS, 256, 0, stream>>>(hidden, HH, 0, w_in, h_norm, HH, HH);
    gemm_bf16_kernel<false><<<dim3(QKVW / 128, SS / 128), 256, 0, stream>>>(
        h_norm, HH, w_qkv, QKVW, qkv, QKVW, SS, QKVW, HH, nullptr);
    rmsnorm_kernel<<<SS, 256, 0, stream>>>(qkv, QKVW, 0, w_inter, q_rms, QQ, QQ);
    gemm_bf16_kernel<false><<<dim3((NH * DD) / 128, SS / 128), 256, 0, stream>>>(
        q_rms, QQ, w_q, NH * DD, q_proj, NH * DD, SS, NH * DD, QQ, nullptr);
    rope_kernel<<<SS * NH, DD / 2, 0, stream>>>(q_proj, DD, 0, NH);
    rope_kernel<<<SS, DD / 2, 0, stream>>>(qkv, QKVW, QQ, 1);
    attn_kernel<<<dim3(SS / 128, NH), 256, 0, stream>>>(q_proj, qkv, attn_out);
    gemm_bf16_kernel<false><<<dim3(HH / 128, SS / 128), 256, 0, stream>>>(
        attn_out, NH * DD, w_o, HH, resid2, HH, SS, HH, NH * DD, hidden);
    rmsnorm_kernel<<<SS, 256, 0, stream>>>(resid2, HH, 0, w_post, h_post, HH, HH);
    glu_gemm_kernel<<<dim3(II / 128, SS / 128, 1), 256, 0, stream>>>(
        h_post, HH, w_gu_se, 0, 2 * II, act_se, nullptr, II, HH, II, SS,
        nullptr, nullptr, nullptr);
    gemm_bf16_kernel<false><<<dim3(HH / 128, SS / 128), 256, 0, stream>>>(
        act_se, II, w_dn_se, HH, share, HH, SS, HH, II, nullptr);
    router_kernel<<<SS / 4, 128, 0, stream>>>(h_post, w_router, topw, topi);
    route_build_kernel<<<1, 256, 0, stream>>>(topi, topw, seg_off, seg_cnt, assign_tok, assign_w);
    zero_kernel<<<(SS * HH + 255) / 256, 256, 0, stream>>>(moe, SS * HH);
    glu_gemm_kernel<<<dim3(II / 128, SS / 128, EE), 256, 0, stream>>>(
        h_post, HH, w_gu_e, (size_t)HH * 2 * II, 2 * II, nullptr, act_e, II, HH, II, 0,
        seg_off, seg_cnt, assign_tok);
    down_moe_kernel<<<dim3(HH / 128, SS / 128, EE), 256, 0, stream>>>(
        act_e, II, w_dn_e, (size_t)II * HH, HH, moe, HH, II,
        seg_off, seg_cnt, assign_tok, assign_w);
    final_add_kernel<<<(SS * HH + 255) / 256, 256, 0, stream>>>(share, moe, resid2, out, SS * HH);

    (void)in_sizes; (void)n_in; (void)out_size; (void)ws_size;
}
